// Decoder_49194555409017
// MI455X (gfx1250) — matrix operand order repaired and checked by reading, compiled, not run on hardware
//
#include <hip/hip_runtime.h>
#include <hip/hip_bf16.h>

// ---------------------------------------------------------------------------
// Transformer decoder forward for MI455X (gfx1250, wave32, WMMA).
// GEMMs and attention both run on v_wmma_f32_16x16x32_bf16 (f32 accum).
// ---------------------------------------------------------------------------

typedef __bf16 bf16;
typedef bf16  v16bf __attribute__((ext_vector_type(16)));
typedef bf16  v8bf  __attribute__((ext_vector_type(8)));
typedef bf16  v4bf  __attribute__((ext_vector_type(4)));
typedef float v8f   __attribute__((ext_vector_type(8)));
typedef float v4f   __attribute__((ext_vector_type(4)));

static __device__ __forceinline__ bf16 f32_to_bf16(float f) {
  unsigned u = __float_as_uint(f);
  unsigned r = (u + 0x7FFFu + ((u >> 16) & 1u)) >> 16;   // round-to-nearest-even
  unsigned short s = (unsigned short)r;
  return *reinterpret_cast<bf16*>(&s);
}

static __device__ __forceinline__ v16bf cat8(v8bf lo, v8bf hi) {
  return __builtin_shufflevector(lo, hi, 0,1,2,3,4,5,6,7,8,9,10,11,12,13,14,15);
}

// Block-wide reduction over 256 threads. op: 0 = sum, 1 = max.
static __device__ __forceinline__ float blk_reduce(float v, float* buf, int op) {
  int tid = threadIdx.x;
  buf[tid] = v;
  __syncthreads();
#pragma unroll
  for (int s = 128; s > 0; s >>= 1) {
    if (tid < s) buf[tid] = op ? fmaxf(buf[tid], buf[tid + s]) : (buf[tid] + buf[tid + s]);
    __syncthreads();
  }
  float r = buf[0];
  __syncthreads();
  return r;
}

// ---------------------------------------------------------------------------
// Generic GEMM:  C[M,N] = act( A[M,K] @ W[K,N] + bias + residual )
// 128x128 block tile, BK=32, 8 waves (4 row x 2 col), 8 WMMA/wave/K-step.
// Register + LDS double buffering: next tile's global loads stay in flight
// across the current tile's WMMA chain (s_wait_loadcnt deferred one step).
// ---------------------------------------------------------------------------
#define BM 128
#define BN 128
#define BK 32
#define LDA 40   // padded bf16 stride (80 B: 16B-aligned, conflict-breaking)
#define LDB 40

__global__ __launch_bounds__(256)
void gemm_bf16_wmma(const float* __restrict__ A, const float* __restrict__ W,
                    const float* __restrict__ bias, const float* __restrict__ res,
                    float* __restrict__ C, int M, int N, int K, int act) {
  __shared__ bf16 As[2][BM * LDA];
  __shared__ bf16 Bs[2][BN * LDB];

  const int tid  = threadIdx.x;
  const int wave = tid >> 5;
  const int lane = tid & 31;
  const int wrow = wave & 3;    // 4 row groups of 32 rows
  const int wcol = wave >> 2;   // 2 col groups of 64 cols
  const int m0 = blockIdx.y * BM;
  const int n0 = blockIdx.x * BN;

  const int halfsel = (lane < 16) ? 0 : 1;   // which K-half this lane supplies
  const int lrow    = lane & 15;

  v8f acc[2][4];
#pragma unroll
  for (int s = 0; s < 2; ++s)
#pragma unroll
    for (int j = 0; j < 4; ++j)
      acc[s][j] = (v8f){0.f,0.f,0.f,0.f,0.f,0.f,0.f,0.f};

  v4f areg[4], breg[4];   // register staging for one A tile + one B tile

  auto glb_load = [&](int k0) {
#pragma unroll
    for (int it = 0; it < 4; ++it) {
      int i  = tid + it * 256;
      int r  = i >> 3;
      int c4 = (i & 7) << 2;
      areg[it] = *(const v4f*)&A[(size_t)(m0 + r) * K + k0 + c4];
    }
#pragma unroll
    for (int it = 0; it < 4; ++it) {
      int i   = tid + it * 256;
      int n   = i & 127;
      int kk4 = (i >> 7) << 2;
#pragma unroll
      for (int j = 0; j < 4; ++j)
        breg[it][j] = W[(size_t)(k0 + kk4 + j) * N + n0 + n];   // coalesced in n
    }
  };

  auto cvt_store = [&](int buf) {
    bf16* Asb = &As[buf][0];
    bf16* Bsb = &Bs[buf][0];
#pragma unroll
    for (int it = 0; it < 4; ++it) {
      int i  = tid + it * 256;
      int r  = i >> 3;
      int c4 = (i & 7) << 2;
      v4bf ap;
#pragma unroll
      for (int j = 0; j < 4; ++j) ap[j] = f32_to_bf16(areg[it][j]);
      *(v4bf*)&Asb[r * LDA + c4] = ap;
    }
#pragma unroll
    for (int it = 0; it < 4; ++it) {
      int i   = tid + it * 256;
      int n   = i & 127;
      int kk4 = (i >> 7) << 2;
      v4bf bp;
#pragma unroll
      for (int j = 0; j < 4; ++j) bp[j] = f32_to_bf16(breg[it][j]);
      *(v4bf*)&Bsb[n * LDB + kk4] = bp;
    }
  };

  const int kTiles = K / BK;
  glb_load(0);

  for (int kt = 0; kt < kTiles; ++kt) {
    const int cur = kt & 1;
    cvt_store(cur);
    __syncthreads();
    if (kt + 1 < kTiles) glb_load((kt + 1) * BK);   // fire-and-forget prefetch

    const bf16* Ac = &As[cur][0];
    const bf16* Bc = &Bs[cur][0];

    v16bf bfrag[4];
#pragma unroll
    for (int j = 0; j < 4; ++j) {
      int bcol = wcol * 64 + j * 16 + lrow;
      bfrag[j] = cat8(*(const v8bf*)&Bc[bcol * LDB + halfsel * 16],
                      *(const v8bf*)&Bc[bcol * LDB + halfsel * 16 + 8]);
    }
#pragma unroll
    for (int s = 0; s < 2; ++s) {
      int arow = wrow * 32 + s * 16 + lrow;
      v16bf afrag = cat8(*(const v8bf*)&Ac[arow * LDA + halfsel * 16],
                         *(const v8bf*)&Ac[arow * LDA + halfsel * 16 + 8]);
#pragma unroll
      for (int j = 0; j < 4; ++j)
        acc[s][j] = __builtin_amdgcn_wmma_f32_16x16x32_bf16(
            false, afrag, false, bfrag[j], (short)0, acc[s][j], false, false);
    }
  }

#pragma unroll
  for (int s = 0; s < 2; ++s) {
#pragma unroll
    for (int j = 0; j < 4; ++j) {
#pragma unroll
      for (int v = 0; v < 8; ++v) {
        int row = m0 + wrow * 32 + s * 16 + v + halfsel * 8;
        int col = n0 + wcol * 64 + j * 16 + lrow;
        float o = acc[s][j][v];
        if (bias) o += bias[col];
        if (res)  o += res[(size_t)row * N + col];
        if (act == 1) o = fmaxf(o, 0.f);
        C[(size_t)row * N + col] = o;
      }
    }
  }
}

// ---------------------------------------------------------------------------
// Flash attention on WMMA. Block = 128 threads (4 waves); each wave owns a
// 16-row q-tile (block covers 64 q). Per 32-wide k-tile:
//   - stage K (32x64) and V^T (64x32) bf16 tiles in LDS (shared by 4 waves)
//   - S^T(32k x 16q) = K-tile @ Q^T  (4 WMMA: 2 k-subtiles x 2 dh-halves)
//     q sits in the lane index, k in the VGPR index -> softmax reductions are
//     per-lane + one __shfl_xor(16).
//   - online softmax (running m,l; rescale ctx accumulators)
//   - ctx^T(64d x 16q) += V^T @ P  (4 WMMA; P fragment built with 2 shfl_xor)
// Causal: k-tiles above the block/wave diagonal are skipped (uniform branch,
// no barriers inside, EXEC all-ones around every WMMA).
// ---------------------------------------------------------------------------
#define LDK 72   // 144 B rows (16B-aligned)
#define LDV 40   // 80 B rows

__global__ __launch_bounds__(128)
void flash_attn_wmma(const float* __restrict__ Q, const float* __restrict__ Kb,
                     const float* __restrict__ Vb, const unsigned char* __restrict__ pad,
                     int Tq, int Tk, int H, int causal, float scale,
                     float* __restrict__ Ctx) {
  __shared__ bf16  Ks[32 * LDK];
  __shared__ bf16  Vt[64 * LDV];
  __shared__ float padv[32];

  const int tid  = threadIdx.x;
  const int wave = tid >> 5;
  const int lane = tid & 31;
  const int lrow = lane & 15;
  const int halfsel = lane >> 4;
  const int h = blockIdx.y, b = blockIdx.z;
  const int hoff  = h * 64;
  const int q0    = blockIdx.x * 64 + wave * 16;
  const int qglob = q0 + lrow;

  // Persistent Q^T B-fragments (dh halves 0..31 / 32..63), pre-scaled bf16.
  v16bf qfrag[2];
#pragma unroll
  for (int half = 0; half < 2; ++half) {
    const float* qr = Q + (size_t)(b * Tq + qglob) * H + hoff + half * 32 + halfsel * 16;
    v16bf f;
#pragma unroll
    for (int i = 0; i < 16; ++i) f[i] = f32_to_bf16(qr[i] * scale);
    qfrag[half] = f;
  }

  v8f acc_c[4];
#pragma unroll
  for (int dt = 0; dt < 4; ++dt)
    acc_c[dt] = (v8f){0.f,0.f,0.f,0.f,0.f,0.f,0.f,0.f};
  float mrun = -3.0e38f, lrun = 0.f;

  const int kTiles = Tk >> 5;
  for (int kt = 0; kt < kTiles; ++kt) {
    const int kbase = kt << 5;
    if (causal && kbase > blockIdx.x * 64 + 63) break;   // block-uniform
    __syncthreads();
    // Stage K tile (32 x 64) row-major bf16.
#pragma unroll
    for (int it = 0; it < 4; ++it) {
      int i  = tid + it * 128;            // 512 vec4 chunks
      int kr = i >> 4, c4 = (i & 15) << 2;
      v4f kv = *(const v4f*)&Kb[(size_t)(b * Tk + kbase + kr) * H + hoff + c4];
      v4bf p;
#pragma unroll
      for (int j = 0; j < 4; ++j) p[j] = f32_to_bf16(kv[j]);
      *(v4bf*)&Ks[kr * LDK + c4] = p;
    }
    // Stage V^T tile (64 x 32): thread owns (d, 4 consecutive k).
#pragma unroll
    for (int it = 0; it < 4; ++it) {
      int i  = tid + it * 128;
      int d  = i & 63, k4 = (i >> 6) << 2;
      v4bf p;
#pragma unroll
      for (int j = 0; j < 4; ++j)
        p[j] = f32_to_bf16(Vb[(size_t)(b * Tk + kbase + k4 + j) * H + hoff + d]);
      *(v4bf*)&Vt[d * LDV + k4] = p;
    }
    if (tid < 32) padv[tid] = pad[b * Tk + kbase + tid] ? -1e18f : 0.f;
    __syncthreads();

    if (causal && kbase > q0 + 15) continue;   // wave-uniform skip (no barriers below)

    // S^T tiles: two 16-k subtiles, K reduced over dh=64 via 2 chained WMMA.
    v8f st[2];
#pragma unroll
    for (int j = 0; j < 2; ++j) {
      v8f s = (v8f){0.f,0.f,0.f,0.f,0.f,0.f,0.f,0.f};
      int row = j * 16 + lrow;
#pragma unroll
      for (int half = 0; half < 2; ++half) {
        int dbase = half * 32;
        v16bf kf = cat8(*(const v8bf*)&Ks[row * LDK + dbase + halfsel * 16],
                        *(const v8bf*)&Ks[row * LDK + dbase + halfsel * 16 + 8]);
        s = __builtin_amdgcn_wmma_f32_16x16x32_bf16(
            false, kf, false, qfrag[half], (short)0, s, false, false);
      }
#pragma unroll
      for (int v = 0; v < 8; ++v) {
        int klocal = j * 16 + halfsel * 8 + v;
        float sv = s[v] + padv[klocal];
        if (causal && (kbase + klocal > qglob)) sv = -1e18f;
        s[v] = sv;
      }
      st[j] = s;
    }

    // Online softmax update; q is per-lane (lrow), k spans VGPRs + half swap.
    float mt = -3.0e38f;
#pragma unroll
    for (int j = 0; j < 2; ++j)
#pragma unroll
      for (int v = 0; v < 8; ++v) mt = fmaxf(mt, st[j][v]);
    mt = fmaxf(mt, __shfl_xor(mt, 16));
    float mnew = fmaxf(mrun, mt);
    float fac  = expf(mrun - mnew);
    float psum = 0.f;
#pragma unroll
    for (int j = 0; j < 2; ++j)
#pragma unroll
      for (int v = 0; v < 8; ++v) {
        float p = expf(st[j][v] - mnew);
        st[j][v] = p;
        psum += p;
      }
    psum += __shfl_xor(psum, 16);
    lrun = lrun * fac + psum;
    mrun = mnew;
#pragma unroll
    for (int dt = 0; dt < 4; ++dt)
#pragma unroll
      for (int v = 0; v < 8; ++v) acc_c[dt][v] *= fac;

    // Build P B-fragment (32k x 16q): lower lanes need st[0] rows {own 0..7,
    // partner 8..15}; upper lanes need st[1] rows {partner 0..7, own 8..15}.
    v16bf pfrag;
#pragma unroll
    for (int v = 0; v < 8; ++v) {
      float sw0 = __shfl_xor(st[0][v], 16);
      float sw1 = __shfl_xor(st[1][v], 16);
      float first = halfsel ? sw1      : st[0][v];
      float last  = halfsel ? st[1][v] : sw0;
      pfrag[v]     = f32_to_bf16(first);
      pfrag[v + 8] = f32_to_bf16(last);
    }

    // ctx^T += V^T @ P (4 d-tiles of 16).
#pragma unroll
    for (int dt = 0; dt < 4; ++dt) {
      int row = dt * 16 + lrow;
      v16bf vf = cat8(*(const v8bf*)&Vt[row * LDV + halfsel * 16],
                      *(const v8bf*)&Vt[row * LDV + halfsel * 16 + 8]);
      acc_c[dt] = __builtin_amdgcn_wmma_f32_16x16x32_bf16(
          false, vf, false, pfrag, (short)0, acc_c[dt], false, false);
    }
  }

  // ctx^T D-layout: lane -> q = lrow; d = dt*16 + v + halfsel*8.
  float invl = 1.0f / lrun;
#pragma unroll
  for (int dt = 0; dt < 4; ++dt)
#pragma unroll
    for (int v = 0; v < 8; ++v) {
      int d = dt * 16 + v + halfsel * 8;
      Ctx[(size_t)(b * Tq + qglob) * H + hoff + d] = acc_c[dt][v] * invl;
    }
}

// ---------------------------------------------------------------------------
// Embedding + sinusoidal positional encoding: x = emb[tgt]*sqrt(H) + PE.
// ---------------------------------------------------------------------------
__global__ __launch_bounds__(256)
void embed_kernel(const int* __restrict__ tgt, const float* __restrict__ emb,
                  float* __restrict__ x, int T, int H) {
  int idx = blockIdx.x * 256 + threadIdx.x;
  int b  = idx / (T * H);
  int r  = idx - b * T * H;
  int t  = r / H;
  int hh = r - t * H;
  int token = tgt[b * T + t];
  float e   = (float)(hh & ~1) / (float)H;
  float ang = (float)t * expf(-9.210340371976184f * e);   // / 10000^e
  float pe  = (hh & 1) ? cosf(ang) : sinf(ang);
  x[idx] = emb[(size_t)token * H + hh] * 32.0f + pe;      // sqrt(1024) = 32
}

// ---------------------------------------------------------------------------
// Row LayerNorm (H = 1024; one block per row, 4 elements per thread).
// ---------------------------------------------------------------------------
__global__ __launch_bounds__(256)
void layernorm_kernel(const float* __restrict__ x, const float* __restrict__ g,
                      const float* __restrict__ b, float* __restrict__ y, int H) {
  __shared__ float red[256];
  int row = blockIdx.x, tid = threadIdx.x;
  const float* xr = x + (size_t)row * H;
  float v[4], s = 0.f;
#pragma unroll
  for (int i = 0; i < 4; ++i) { v[i] = xr[tid + i * 256]; s += v[i]; }
  float mean = blk_reduce(s, red, 0) / (float)H;
  float sq = 0.f;
#pragma unroll
  for (int i = 0; i < 4; ++i) { float d = v[i] - mean; sq += d * d; }
  float var = blk_reduce(sq, red, 0) / (float)H;
  float inv = rsqrtf(var + 1e-6f);
  float* yr = y + (size_t)row * H;
#pragma unroll
  for (int i = 0; i < 4; ++i) {
    int col = tid + i * 256;
    yr[col] = (v[i] - mean) * inv * g[col] + b[col];
  }
}

// Save layer input into result2 slot: out2[b, l, t, h] = x[b, t, h].
__global__ __launch_bounds__(256)
void save_kernel(const float* __restrict__ x, float* __restrict__ out2,
                 int l, int L, int TH) {
  int idx = blockIdx.x * 256 + threadIdx.x;
  int b = idx / TH;
  int r = idx - b * TH;
  out2[((size_t)b * L + l) * TH + r] = x[idx];
}

// ---------------------------------------------------------------------------
// ws[-1]: recompute last-layer cross-attention weights, averaged over heads.
// One block per (b, q); each thread owns 4 k-positions (S = 1024).
// ---------------------------------------------------------------------------
__global__ __launch_bounds__(256)
void cross_avg_kernel(const float* __restrict__ Q, const float* __restrict__ Kb,
                      const unsigned char* __restrict__ pad, float* __restrict__ out,
                      int Tq, int Tk, int H, float scale) {
  __shared__ float qs[64];
  __shared__ float red[256];
  const int q = blockIdx.x, b = blockIdx.y, tid = threadIdx.x;
  float accw[4] = {0.f, 0.f, 0.f, 0.f};

  for (int h = 0; h < 16; ++h) {
    __syncthreads();
    if (tid < 64) qs[tid] = Q[((size_t)(b * Tq + q) * H) + h * 64 + tid] * scale;
    __syncthreads();
    float s[4], lmax = -3.0e38f;
#pragma unroll
    for (int i = 0; i < 4; ++i) {
      int k = tid + (i << 8);
      const v4f* kr = (const v4f*)(Kb + ((size_t)(b * Tk + k) * H) + h * 64);
      float acc = 0.f;
#pragma unroll
      for (int d4 = 0; d4 < 16; ++d4) {
        v4f kv = kr[d4];
        acc += qs[d4 * 4 + 0] * kv[0] + qs[d4 * 4 + 1] * kv[1]
             + qs[d4 * 4 + 2] * kv[2] + qs[d4 * 4 + 3] * kv[3];
      }
      if (pad[b * Tk + k]) acc = -1e18f;
      s[i] = acc;
      lmax = fmaxf(lmax, acc);
    }
    float m = blk_reduce(lmax, red, 1);
    float lsum = 0.f;
#pragma unroll
    for (int i = 0; i < 4; ++i) { s[i] = expf(s[i] - m); lsum += s[i]; }
    float inv = 1.0f / blk_reduce(lsum, red, 0);
#pragma unroll
    for (int i = 0; i < 4; ++i) accw[i] += s[i] * inv * 0.0625f;   // / NH
  }
#pragma unroll
  for (int i = 0; i < 4; ++i)
    out[((size_t)(b * Tq + q) * Tk) + tid + (i << 8)] = accw[i];
}

// p_trans: sigmoid( t1_row . Wt2 ), one block per row.
__global__ __launch_bounds__(256)
void sigdot_kernel(const float* __restrict__ t1, const float* __restrict__ Wt2,
                   float* __restrict__ out, int H) {
  __shared__ float red[256];
  int row = blockIdx.x, tid = threadIdx.x;
  float s = 0.f;
  for (int i = tid; i < H; i += 256) s += t1[(size_t)row * H + i] * Wt2[i];
  float tot = blk_reduce(s, red, 0);
  if (tid == 0) out[row] = 1.0f / (1.0f + expf(-tot));
}

// ---------------------------------------------------------------------------
// Host orchestration
// ---------------------------------------------------------------------------
extern "C" void kernel_launch(void* const* d_in, const int* in_sizes, int n_in,
                              void* d_out, int out_size, void* d_ws, size_t ws_size,
                              hipStream_t stream) {
  (void)in_sizes; (void)n_in; (void)out_size; (void)ws_size;
  const int B = 4, T = 512, S = 1024, H = 1024, F = 4096, Lc = 6, NHc = 16;

  const int*           tgt     = (const int*)d_in[0];
  const unsigned char* src_pad = (const unsigned char*)d_in[1];
  const unsigned char* tgt_pad = (const unsigned char*)d_in[2];
  const float* enc     = (const float*)d_in[3];
  const float* emb     = (const float*)d_in[4];
  const float* Wq_self = (const float*)d_in[5];
  const float* Wk_self = (const float*)d_in[6];
  const float* Wv_self = (const float*)d_in[7];
  const float* Wo_self = (const float*)d_in[8];
  const float* bo_self = (const float*)d_in[9];
  const float* Wq_src  = (const float*)d_in[10];
  const float* Wk_src  = (const float*)d_in[11];
  const float* Wv_src  = (const float*)d_in[12];
  const float* Wo_src  = (const float*)d_in[13];
  const float* bo_src  = (const float*)d_in[14];
  const float* W1      = (const float*)d_in[15];
  const float* W2      = (const float*)d_in[16];
  const float* ln_g    = (const float*)d_in[17];
  const float* ln_b    = (const float*)d_in[18];
  const float* final_g = (const float*)d_in[19];
  const float* final_b = (const float*)d_in[20];
  const float* Wt1     = (const float*)d_in[21];
  const float* Wt2     = (const float*)d_in[22];

  float* out = (float*)d_out;
  float* r1    = out;                                 // (B,T,H)
  float* r2    = out + (size_t)B * T * H;             // (B,L,T,H)
  float* ptOut = out + (size_t)B * T * H * (1 + Lc);  // (B,T,1)
  float* wsOut = ptOut + (size_t)B * T;               // (B,T,S)

  float* w    = (float*)d_ws;                 // ~104 MB of f32 scratch
  float* x    = w;
  float* hbuf = x    + (size_t)B * T * H;
  float* qb   = hbuf + (size_t)B * T * H;
  float* kb   = qb   + (size_t)B * T * H;
  float* vb   = kb   + (size_t)B * S * H;
  float* ctx  = vb   + (size_t)B * S * H;
  float* mid  = ctx  + (size_t)B * T * H;
  float* t1   = mid  + (size_t)B * T * F;

  auto gemm = [&](const float* A, const float* Wm, const float* bias,
                  const float* res, float* C, int M, int N, int K, int act) {
    gemm_bf16_wmma<<<dim3(N / BN, M / BM), 256, 0, stream>>>(A, Wm, bias, res, C, M, N, K, act);
  };

  embed_kernel<<<(B * T * H) / 256, 256, 0, stream>>>(tgt, emb, x, T, H);

  const size_t HH = (size_t)H * H;
  for (int l = 0; l < Lc; ++l) {
    save_kernel<<<(B * T * H) / 256, 256, 0, stream>>>(x, r2, l, Lc, T * H);

    // --- self attention ---
    layernorm_kernel<<<B * T, 256, 0, stream>>>(x, ln_g + (size_t)(l * 3 + 0) * H,
                                                ln_b + (size_t)(l * 3 + 0) * H, hbuf, H);
    gemm(hbuf, Wq_self + l * HH, nullptr, nullptr, qb, B * T, H, H, 0);
    gemm(hbuf, Wk_self + l * HH, nullptr, nullptr, kb, B * T, H, H, 0);
    gemm(hbuf, Wv_self + l * HH, nullptr, nullptr, vb, B * T, H, H, 0);
    flash_attn_wmma<<<dim3(T / 64, NHc, B), 128, 0, stream>>>(qb, kb, vb, tgt_pad,
                                                              T, T, H, 1, 0.125f, ctx);
    gemm(ctx, Wo_self + l * HH, bo_self + (size_t)l * H, x, x, B * T, H, H, 0);

    // --- cross attention ---
    layernorm_kernel<<<B * T, 256, 0, stream>>>(x, ln_g + (size_t)(l * 3 + 1) * H,
                                                ln_b + (size_t)(l * 3 + 1) * H, hbuf, H);
    gemm(hbuf, Wq_src + l * HH, nullptr, nullptr, qb, B * T, H, H, 0);
    gemm(enc,  Wk_src + l * HH, nullptr, nullptr, kb, B * S, H, H, 0);
    gemm(enc,  Wv_src + l * HH, nullptr, nullptr, vb, B * S, H, H, 0);
    flash_attn_wmma<<<dim3(T / 64, NHc, B), 128, 0, stream>>>(qb, kb, vb, src_pad,
                                                              T, S, H, 0, 0.125f, ctx);
    gemm(ctx, Wo_src + l * HH, bo_src + (size_t)l * H, x, x, B * T, H, H, 0);

    // --- FFN ---
    layernorm_kernel<<<B * T, 256, 0, stream>>>(x, ln_g + (size_t)(l * 3 + 2) * H,
                                                ln_b + (size_t)(l * 3 + 2) * H, hbuf, H);
    gemm(hbuf, W1 + (size_t)l * H * F, nullptr, nullptr, mid, B * T, F, H, 1);
    gemm(mid,  W2 + (size_t)l * F * H, nullptr, x, x, B * T, H, F, 0);
  }

  layernorm_kernel<<<B * T, 256, 0, stream>>>(x, final_g, final_b, r1, H);
  gemm(x, Wt1, nullptr, nullptr, t1, B * T, H, H, 0);
  sigdot_kernel<<<B * T, 256, 0, stream>>>(t1, Wt2, ptOut, H);
  // q/k of last layer's cross attention are still live in qb/kb.
  cross_avg_kernel<<<dim3(T, B), 256, 0, stream>>>(qb, kb, src_pad, wsOut, T, S, H, 0.125f);
}
